// CaserModel_38053410243179
// MI455X (gfx1250) — matrix operand order repaired and checked by reading, not compiled, not run on hardware
//
#include <hip/hip_runtime.h>

#define B_TOT 8192
#define SL    8      // L
#define DD    128    // D
#define FV    64
#define FH    64
#define FC_IN (FV*DD + FH*SL)   // 8704
#define TB    16                 // batch rows per workgroup
#define NCHUNK (FC_IN/32)        // 272 K-chunks of 32 for the FC
#define NUM_ITEMS_C 100000

typedef __attribute__((ext_vector_type(16))) __bf16 v16bf;
typedef __attribute__((ext_vector_type(8)))  __bf16 v8bf;
typedef __attribute__((ext_vector_type(8)))  float  v8f;
typedef __attribute__((ext_vector_type(4)))  int    v4i;

// ---- addrspace helpers (integer round-trip; generic-LDS addr[31:0] == LDS offset) ----
typedef __attribute__((address_space(1))) v4i gv4i;   // global 16-byte block
typedef __attribute__((address_space(3))) v4i lv4i;   // LDS 16-byte block
static __device__ inline gv4i* to_g128(const void* p) {
  return (gv4i*)(unsigned long long)p;
}
static __device__ inline lv4i* to_l128(void* p) {
  return (lv4i*)(unsigned int)(unsigned long long)p;
}

#if defined(__AMDGCN__) && __has_builtin(__builtin_amdgcn_global_load_async_to_lds_b128)
#define HAVE_ASYNC_LDS 1
#else
#define HAVE_ASYNC_LDS 0
#endif

static __device__ inline void wait_async_zero() {
#if __has_builtin(__builtin_amdgcn_s_wait_asynccnt)
  __builtin_amdgcn_s_wait_asynccnt(0);
#else
  asm volatile("s_wait_asynccnt 0x0" ::: "memory");
#endif
}

// Build a 16-element bf16 WMMA fragment from two 16-byte chunks.
static __device__ inline v16bf make_frag(const __bf16* p0, const __bf16* p1) {
  v8bf lo = *(const v8bf*)p0;
  v8bf hi = *(const v8bf*)p1;
  v16bf r;
#pragma unroll
  for (int i = 0; i < 8; ++i) { r[i] = lo[i]; r[i+8] = hi[i]; }
  return r;
}

// A-matrix (16x32 bf16, row-major source, 32-elem chunk base `row`):
// lane = M + 16*half ; element e -> K = e + 8*half + 8*(e>=8)
static __device__ inline v16bf load_a_frag(const __bf16* row, int half) {
  return make_frag(row + half*8, row + 16 + half*8);
}
// B-matrix (32x16 bf16 from row-major [N][K], chunk base `row` = &W[N][c*32]):
// lane = N + 16*half ; element e -> K = e + 8*half + 8*(e>=8)
static __device__ inline v16bf load_b_frag(const __bf16* row, int half) {
  return make_frag(row + half*8, row + 16 + half*8);
}

__global__ __launch_bounds__(256)
void cvt_f32_to_bf16(const float* __restrict__ src, __bf16* __restrict__ dst, int n) {
  int i = blockIdx.x * 256 + threadIdx.x;
  int stride = gridDim.x * 256;
  for (; i < n; i += stride) dst[i] = (__bf16)src[i];
}

__global__ __launch_bounds__(256)
void caser_fused(const int* __restrict__ user_id, const int* __restrict__ seq_L,
                 const int* __restrict__ item_id,
                 const __bf16* __restrict__ Qbf,   // [NUM_ITEMS][128] bf16 (ws)
                 const float* __restrict__ P,
                 const float* __restrict__ Qp, const float* __restrict__ b_item,
                 const float* __restrict__ Wv, const float* __restrict__ bv,
                 const float* __restrict__ bh, const float* __restrict__ bfc,
                 const __bf16* __restrict__ WfcB,  // [128][8704] bf16 (ws)
                 const __bf16* __restrict__ WhB,   // [8][64][8][128] bf16 (ws)
                 float* __restrict__ out)
{
  extern __shared__ __align__(16) char smem[];
  __bf16* x_lds  = (__bf16*)smem;                    // 16*8*128 el   = 32768 B
  __bf16* comb   = (__bf16*)(smem + 32768);          // 16*8704 el    = 278528 B
  float*  z_lds  = (float*)smem;                     // reuses x region after phase 2
  float*  psum   = (float*)(smem + 8192);            // 256 floats (inside old x region)
  float*  wv_lds = (float*)(smem + 32768 + 278528);  // 512 f32
  float*  bv_lds = wv_lds + FV*SL;                   // 64 f32

  const int tid  = threadIdx.x;
  const int wave = tid >> 5;
  const int lane = tid & 31;
  const int mrow = lane & 15;    // M / N index within a fragment
  const int half = lane >> 4;
  const int b0   = blockIdx.x * TB;

  // ---- Phase 1: stage small weights + gather x tile (bf16) ----
  for (int i = tid; i < FV*SL; i += 256) wv_lds[i] = Wv[i];
  for (int i = tid; i < FV;    i += 256) bv_lds[i] = bv[i];
#if HAVE_ASYNC_LDS
  // Async DMA: 2048 16-byte blocks (16 rows x 8 items x 256 B), 8 per thread.
  for (int k = tid; k < TB*SL*DD*2/16; k += 256) {
    int r = k >> 4;              // row = (m,l); 16 blocks of 16 B per 256 B row
    int m = r >> 3, l = r & 7;
    int item = seq_L[(b0 + m)*SL + l];
    const __bf16* gsrc = Qbf + (size_t)item*DD + (k & 15)*8;  // 8 bf16 = 16 B
    __builtin_amdgcn_global_load_async_to_lds_b128(to_g128(gsrc),
                                                   to_l128(x_lds + k*8), 0, 0);
  }
  wait_async_zero();
#else
  for (int i = tid; i < TB*SL*DD; i += 256) {
    int m = i >> 10;            // /(SL*DD) = /1024
    int l = (i >> 7) & 7;
    int d = i & 127;
    int item = seq_L[(b0 + m)*SL + l];
    x_lds[i] = Qbf[(size_t)item*DD + d];
  }
#endif
  __syncthreads();

  // ---- Phase 2a: vertical conv (VALU, cheap) -> comb[:, 0:8192] ----
  for (int p = tid; p < TB*DD; p += 256) {
    int m = p >> 7, d = p & 127;
    float xv[SL];
#pragma unroll
    for (int l = 0; l < SL; ++l) xv[l] = (float)x_lds[(m*SL + l)*DD + d];
    for (int f = 0; f < FV; ++f) {
      float acc = bv_lds[f];
#pragma unroll
      for (int l = 0; l < SL; ++l) acc += wv_lds[f*SL + l] * xv[l];
      comb[m*FC_IN + f*DD + d] = (__bf16)fmaxf(acc, 0.f);
    }
  }

  // ---- Phase 2b: horizontal convs via WMMA -> comb[:, 8192:8704] ----
  // 32 (h, n-tile) tasks round-robined over 8 waves.
  for (int task = wave; task < 32; task += 8) {
    int h  = (task >> 2) + 1;          // 1..8
    int nt = task & 3;                 // filter tile (64 filters / 16)
    int f  = nt*16 + mrow;             // this lane's N (filter) index
    float vmax[8];
#pragma unroll
    for (int i = 0; i < 8; ++i) vmax[i] = -3.0e38f;
    for (int j = 0; j <= SL - h; ++j) {       // window positions
      v8f acc = {};
      for (int c = 0; c < 4*h; ++c) {         // K = h*128, chunks of 32
        int hh    = c >> 2;
        int dbase = (c & 3) * 32;
        const __bf16* arow = &x_lds[(mrow*SL + j + hh)*DD + dbase];
        v16bf a  = load_a_frag(arow, half);
        const __bf16* brow = &WhB[(((h-1)*FH + f)*SL + hh)*DD + dbase];
        v16bf bm = load_b_frag(brow, half);
        acc = __builtin_amdgcn_wmma_f32_16x16x32_bf16(false, a, false, bm,
                                                      (short)0, acc, false, false);
      }
#pragma unroll
      for (int i = 0; i < 8; ++i) vmax[i] = fmaxf(vmax[i], acc[i]);
    }
    float bhv = bh[(h-1)*FH + f];
#pragma unroll
    for (int i = 0; i < 8; ++i) {
      int M = i + half*8;
      comb[M*FC_IN + FV*DD + (h-1)*FH + f] = (__bf16)fmaxf(vmax[i] + bhv, 0.f);
    }
  }
  __syncthreads();

  // ---- Phase 3: FC (16x8704 @ 8704x128) via WMMA; wave w owns N-tile w ----
  {
    const int nt = wave;  // 8 waves x 16 cols = 128 outputs
    const __bf16* bbase = &WfcB[(size_t)(nt*16 + mrow)*FC_IN];
    v8f acc = {};
    for (int c = 0; c < NCHUNK; ++c) {
      if (c + 4 < NCHUNK) __builtin_prefetch(bbase + (c+4)*32, 0, 1);
      const __bf16* arow = &comb[mrow*FC_IN + c*32];
      v16bf a  = load_a_frag(arow, half);
      v16bf bm = load_b_frag(bbase + c*32, half);
      acc = __builtin_amdgcn_wmma_f32_16x16x32_bf16(false, a, false, bm,
                                                    (short)0, acc, false, false);
    }
    float bias = bfc[nt*16 + mrow];
#pragma unroll
    for (int i = 0; i < 8; ++i) {
      int M = i + half*8;
      z_lds[M*DD + nt*16 + mrow] = fmaxf(acc[i] + bias, 0.f);
    }
  }
  __syncthreads();

  // ---- Phase 4: score = dot(concat(z, P[user]), Qp[item]) + b_item ----
  {
    int m    = tid >> 4;
    int part = tid & 15;     // 16 lanes x 16 dims = 256
    int u  = user_id[b0 + m];
    int it = item_id[b0 + m];
    float s = 0.f;
#pragma unroll
    for (int k = 0; k < 16; ++k) {
      int d = part*16 + k;
      float xv = (d < DD) ? z_lds[m*DD + d] : P[u*DD + (d - DD)];
      s += xv * Qp[it*(2*DD) + d];
    }
    psum[tid] = s;
  }
  __syncthreads();
  if (tid < TB) {
    int it = item_id[b0 + tid];
    float s = b_item[it];
#pragma unroll
    for (int k = 0; k < 16; ++k) s += psum[tid*16 + k];
    out[b0 + tid] = s;
  }
}

extern "C" void kernel_launch(void* const* d_in, const int* in_sizes, int n_in,
                              void* d_out, int out_size, void* d_ws, size_t ws_size,
                              hipStream_t stream) {
  (void)in_sizes; (void)n_in; (void)out_size; (void)ws_size;
  const int*   user_id = (const int*)d_in[0];
  const int*   seq_L   = (const int*)d_in[1];
  const int*   item_id = (const int*)d_in[2];
  const float* Q       = (const float*)d_in[3];
  const float* P       = (const float*)d_in[4];
  const float* Qp      = (const float*)d_in[5];
  const float* b_item  = (const float*)d_in[6];
  const float* Wv      = (const float*)d_in[7];
  const float* bv      = (const float*)d_in[8];
  const float* Wh      = (const float*)d_in[9];
  const float* bh      = (const float*)d_in[10];
  const float* Wfc     = (const float*)d_in[11];
  const float* bfc     = (const float*)d_in[12];

  const int nQ   = NUM_ITEMS_C * DD;    // 12,800,000
  const int nWfc = DD * FC_IN;          // 1,114,112
  const int nWh  = SL * FH * SL * DD;   // 524,288
  __bf16* q_bf   = (__bf16*)d_ws;
  __bf16* wfc_bf = (__bf16*)((char*)d_ws + (size_t)nQ * 2);
  __bf16* wh_bf  = (__bf16*)((char*)d_ws + (size_t)(nQ + nWfc) * 2);

  cvt_f32_to_bf16<<<4096, 256, 0, stream>>>(Q,   q_bf,   nQ);
  cvt_f32_to_bf16<<<512,  256, 0, stream>>>(Wfc, wfc_bf, nWfc);
  cvt_f32_to_bf16<<<512,  256, 0, stream>>>(Wh,  wh_bf,  nWh);

  // LDS: x(32768) + comb(278528) + Wv/bv f32 (2304) = 313600 B (< 320 KB WGP LDS)
  size_t lds_bytes = 32768 + 278528 + (FV*SL + FV) * sizeof(float);
  caser_fused<<<B_TOT / TB, 256, lds_bytes, stream>>>(
      user_id, seq_L, item_id, q_bf, P, Qp, b_item, Wv, bv, bh, bfc,
      wfc_bf, wh_bf, (float*)d_out);
}
